// Att_MambaLayer_12034498363969
// MI455X (gfx1250) — matrix operand order repaired and checked by reading, not compiled, not run on hardware
//
#include <hip/hip_runtime.h>

// ---------------------------------------------------------------------------
// CDNA5 (gfx1250) Attention+Mamba block.  GEMM-feeding tensors kept in bf16
// end-to-end; all matrix math through v_wmma_f32_16x16x32_bf16 (wave32, f32
// accum).  Fragments load as raw b128 bf16 runs -> zero convert VALU in-loop.
// ---------------------------------------------------------------------------

typedef __attribute__((ext_vector_type(16))) __bf16 v16bf;
typedef __attribute__((ext_vector_type(8)))  float  v8f;
typedef unsigned short u16;

#define DEVINL __device__ __forceinline__

union BfFrag { v16bf v; unsigned u[8]; u16 s[16]; };

DEVINL unsigned bfr(float f) {               // bf16 RNE in high 16 bits
  unsigned u = __float_as_uint(f);
  return u + 0x7fffu + ((u >> 16) & 1u);
}
DEVINL u16 f2bf(float f) { return (u16)(bfr(f) >> 16); }
DEVINL unsigned pk(float lo, float hi) {
  return __builtin_amdgcn_perm(bfr(hi), bfr(lo), 0x07060302u);
}
DEVINL uint4 ldbf8(const u16* p, bool ok) {  // 8 contiguous bf16
  if (ok) return *(const uint4*)p;
  uint4 z; z.x = z.y = z.z = z.w = 0u; return z;
}
DEVINL float4 ld4(const float* p, bool ok) {
  if (ok) return *(const float4*)p;
  float4 z; z.x = z.y = z.z = z.w = 0.f; return z;
}
DEVINL void pack8(BfFrag& f, int base, float4 a, float4 b) {
  f.u[base + 0] = pk(a.x, a.y); f.u[base + 1] = pk(a.z, a.w);
  f.u[base + 2] = pk(b.x, b.y); f.u[base + 3] = pk(b.z, b.w);
}
DEVINL void put8(BfFrag& f, int base, uint4 a) {
  f.u[base + 0] = a.x; f.u[base + 1] = a.y; f.u[base + 2] = a.z; f.u[base + 3] = a.w;
}

// ---------------------------------------------------------------------------
// Strided batched WMMA GEMM.  Wave: 16x32 output (2 WMMAs share A frag);
// block: 8 waves -> 64x64 tile.  A is always bf16 with unit K-stride;
// B is bf16 (unit K-stride) or fp32 (dt GEMM).  Output fp32 or bf16.
//   A[m,k] = A[off + m*sam + k],  B[k,n] = B[off + n*sbn + k]
//   off(z) = (z/zdiv)*b?1 + (z%zdiv)*b?2  (elements)
// NGUARD: column guard (x_dbl N=48).  KGUARD: K-group guard (dt K=16).
// biasMode: 0 none, 1 per-N, 2 per-M.  act: 0 none, 1 relu, 2 softplus.
// ---------------------------------------------------------------------------
struct GemmP {
  const u16* A; const void* B; const float* bias; void* C;
  int M, N, K;
  long sam, sbn, scm, scn;
  int zdiv;
  long bA1, bA2, bB1, bB2, bC1, bC2, bBi1, bBi2;
  int biasMode, act;
};

template <bool BF_B, bool BF_OUT, bool NGUARD, bool KGUARD>
__global__ __launch_bounds__(256) void gemm_wmma(GemmP p) {
  const int lane = threadIdx.x;                 // wave32
  const int wv   = threadIdx.y;                 // 8 waves: 4 in M x 2 in N
  const int half = lane >> 4, l16 = lane & 15;
  const int tm = (blockIdx.y * 4 + (wv >> 1)) * 16;
  const int tn = (blockIdx.x * 2 + (wv & 1)) * 32;
  const int z  = blockIdx.z;
  const long zh = z / p.zdiv, zl = z % p.zdiv;
  const u16*   Ab  = p.A + zh * p.bA1 + zl * p.bA2;
  const u16*   Bbb = (const u16*)p.B + zh * p.bB1 + zl * p.bB2;
  const float* Bbf = (const float*)p.B + zh * p.bB1 + zl * p.bB2;
  const float* biasp = p.bias + zh * p.bBi1 + zl * p.bBi2;

  const int row  = tm + l16;
  const int col0 = tn + l16, col1 = col0 + 16;
  const bool c0ok = !NGUARD || col0 < p.N;
  const bool c1ok = !NGUARD || col1 < p.N;

  v8f acc0 = {0.f,0.f,0.f,0.f,0.f,0.f,0.f,0.f};
  v8f acc1 = {0.f,0.f,0.f,0.f,0.f,0.f,0.f,0.f};

  for (int k0 = 0; k0 < p.K; k0 += 32) {
    BfFrag af, bf0, bf1;
    {   // A: two 8-wide contiguous bf16 runs per lane (contiguous k = 16*half + s, same order as B)
      const u16* ap = Ab + (long)row * p.sam + (k0 + half * 16);
      put8(af, 0, ldbf8(ap,      !KGUARD || (k0 + half * 16)      < p.K));
      put8(af, 4, ldbf8(ap + 8,  !KGUARD || (k0 + half * 16 + 8)  < p.K));
    }
    const int kb = k0 + 16 * half;
    if (BF_B) {   // 16 contiguous bf16 per lane
      const u16* bp0 = Bbb + (long)col0 * p.sbn + kb;
      const u16* bp1 = Bbb + (long)col1 * p.sbn + kb;
      put8(bf0, 0, ldbf8(bp0,     c0ok && (!KGUARD || kb     < p.K)));
      put8(bf0, 4, ldbf8(bp0 + 8, c0ok && (!KGUARD || kb + 8 < p.K)));
      put8(bf1, 0, ldbf8(bp1,     c1ok && (!KGUARD || kb     < p.K)));
      put8(bf1, 4, ldbf8(bp1 + 8, c1ok && (!KGUARD || kb + 8 < p.K)));
    } else {      // fp32 B (dt GEMM), convert via v_perm packing
      const float* bp0 = Bbf + (long)col0 * p.sbn + kb;
      const float* bp1 = Bbf + (long)col1 * p.sbn + kb;
      float4 b0 = ld4(bp0 + 0,  c0ok && (!KGUARD || kb      < p.K));
      float4 b1 = ld4(bp0 + 4,  c0ok && (!KGUARD || kb +  4 < p.K));
      float4 b2 = ld4(bp0 + 8,  c0ok && (!KGUARD || kb +  8 < p.K));
      float4 b3 = ld4(bp0 + 12, c0ok && (!KGUARD || kb + 12 < p.K));
      pack8(bf0, 0, b0, b1); pack8(bf0, 4, b2, b3);
      float4 c0 = ld4(bp1 + 0,  c1ok && (!KGUARD || kb      < p.K));
      float4 c1 = ld4(bp1 + 4,  c1ok && (!KGUARD || kb +  4 < p.K));
      float4 c2 = ld4(bp1 + 8,  c1ok && (!KGUARD || kb +  8 < p.K));
      float4 c3 = ld4(bp1 + 12, c1ok && (!KGUARD || kb + 12 < p.K));
      pack8(bf1, 0, c0, c1); pack8(bf1, 4, c2, c3);
    }
    acc0 = __builtin_amdgcn_wmma_f32_16x16x32_bf16(false, af.v, false, bf0.v,
                                                   (short)0, acc0, false, false);
    acc1 = __builtin_amdgcn_wmma_f32_16x16x32_bf16(false, af.v, false, bf1.v,
                                                   (short)0, acc1, false, false);
  }

  float* Cf = (float*)p.C + zh * p.bC1 + zl * p.bC2;
  u16*   Cs = (u16*)p.C   + zh * p.bC1 + zl * p.bC2;
#pragma unroll
  for (int r = 0; r < 8; ++r) {
    int m = tm + r + half * 8;                    // C: m = r + 8*half
    float v0 = acc0[r], v1 = acc1[r];
    if (p.biasMode == 1)      { v0 += biasp[col0]; if (c1ok) v1 += biasp[col1]; }
    else if (p.biasMode == 2) { float bm = biasp[m]; v0 += bm; v1 += bm; }
    if (p.act == 1)      { v0 = fmaxf(v0, 0.f); v1 = fmaxf(v1, 0.f); }
    else if (p.act == 2) {
      v0 = (v0 > 20.f) ? v0 : log1pf(__expf(v0));
      v1 = (v1 > 20.f) ? v1 : log1pf(__expf(v1));
    }
    long i0 = (long)m * p.scm + (long)col0 * p.scn;
    long i1 = (long)m * p.scm + (long)col1 * p.scn;
    if (BF_OUT) {
      if (c0ok) Cs[i0] = f2bf(v0);
      if (c1ok) Cs[i1] = f2bf(v1);
    } else {
      if (c0ok) Cf[i0] = v0;
      if (c1ok) Cf[i1] = v1;
    }
  }
}

// ---------------------------------------------------------------------------
// Implicit-GEMM 3x3 conv (pad=1), 32x32 imgs, Ci=Co=256, bf16 token-major in,
// bf16 repacked weights; K = 9*256 tap-major.
// ---------------------------------------------------------------------------
struct ConvP { const u16* in; const u16* bw; const float* bias; void* out; int relu; };

template <bool BF_OUT>
__global__ __launch_bounds__(256) void conv3_wmma(ConvP p) {
  const int lane = threadIdx.x, wv = threadIdx.y;
  const int half = lane >> 4, l16 = lane & 15;
  const int tm = (blockIdx.y * 4 + (wv >> 1)) * 16;
  const int tn = (blockIdx.x * 2 + (wv & 1)) * 32;
  const int row = tm + l16, col0 = tn + l16, col1 = col0 + 16;
  const int b = row >> 10, pix = row & 1023;
  const int h = pix >> 5, w = pix & 31;
  const u16* inb = p.in + (long)b * 262144;

  v8f acc0 = {0.f,0.f,0.f,0.f,0.f,0.f,0.f,0.f};
  v8f acc1 = {0.f,0.f,0.f,0.f,0.f,0.f,0.f,0.f};
  for (int k0 = 0; k0 < 2304; k0 += 32) {
    BfFrag af, bf0, bf1;
#pragma unroll
    for (int g = 0; g < 2; ++g) {               // two 8-wide ci runs
      int kk = k0 + half * 16 + g * 8;
      int tap = kk >> 8, ci = kk & 255;
      int hh = h + tap / 3 - 1, ww = w + tap % 3 - 1;
      bool ok = ((unsigned)hh < 32u) && ((unsigned)ww < 32u);
      put8(af, g * 4, ldbf8(inb + (long)(hh * 32 + ww) * 256 + ci, ok));
    }
    const int kb = k0 + 16 * half;
    const u16* bp0 = p.bw + (long)col0 * 2304 + kb;
    const u16* bp1 = p.bw + (long)col1 * 2304 + kb;
    put8(bf0, 0, *(const uint4*)bp0); put8(bf0, 4, *(const uint4*)(bp0 + 8));
    put8(bf1, 0, *(const uint4*)bp1); put8(bf1, 4, *(const uint4*)(bp1 + 8));
    acc0 = __builtin_amdgcn_wmma_f32_16x16x32_bf16(false, af.v, false, bf0.v,
                                                   (short)0, acc0, false, false);
    acc1 = __builtin_amdgcn_wmma_f32_16x16x32_bf16(false, af.v, false, bf1.v,
                                                   (short)0, acc1, false, false);
  }
#pragma unroll
  for (int r = 0; r < 8; ++r) {
    int m = tm + r + half * 8;
    float v0 = acc0[r] + p.bias[col0];
    float v1 = acc1[r] + p.bias[col1];
    if (p.relu) { v0 = fmaxf(v0, 0.f); v1 = fmaxf(v1, 0.f); }
    if (BF_OUT) {
      ((u16*)p.out)[(long)m * 256 + col0] = f2bf(v0);
      ((u16*)p.out)[(long)m * 256 + col1] = f2bf(v1);
    } else {
      ((float*)p.out)[(long)m * 256 + col0] = v0;
      ((float*)p.out)[(long)m * 256 + col1] = v1;
    }
  }
}

// ---------------------------------------------------------------------------
// Elementwise / permute / normalization kernels
// ---------------------------------------------------------------------------

// generic f32 -> bf16 convert (weights)
__global__ void k_f2bfv(const float* __restrict__ s, u16* __restrict__ d, int n) {
  int i = blockIdx.x * 256 + threadIdx.x;
  if (i < n) d[i] = f2bf(s[i]);
}

// conv weight (Co,Ci,3,3) -> bf16 im2col-ordered (Co, 2304), k = tap*256+ci
__global__ void k_wrepack(const float* __restrict__ wt, u16* __restrict__ bw) {
  int i = blockIdx.x * 256 + threadIdx.x;       // 589,824
  int co = i / 2304, k = i - co * 2304;
  int tap = k >> 8, ci = k & 255;
  bw[i] = f2bf(wt[co * 2304 + ci * 9 + tap]);
}

// x_tok[b, pix, c2] (bf16) = (transpose(x,(0,2,1,3)).reshape)[b,c2,pix]
__global__ void k_permute_in(const float* __restrict__ src, u16* __restrict__ dst) {
  int i = blockIdx.x * 256 + threadIdx.x;       // 1,048,576
  int b = i >> 18, r = i & 262143;
  int pix = r >> 8, c2 = r & 255;
  int j = (c2 << 10) + pix;
  int h = j >> 13, c = (j >> 5) & 255, w = j & 31;
  dst[i] = f2bf(src[(b << 18) + (c << 10) + (h << 5) + w]);
}

// hs[b, l, d] (bf16) = atto[b].flat[d*1024 + l]
__global__ void k_scramble(const float* __restrict__ src, u16* __restrict__ dst) {
  int i = blockIdx.x * 256 + threadIdx.x;       // 1,048,576
  int j = i & 262143;                            // j = l*256 + d
  dst[i] = f2bf(src[(i - j) + ((j & 255) << 10) + (j >> 8)]);
}

__global__ void k_reverse(const float* __restrict__ src, float* __restrict__ dst) {
  int i = blockIdx.x * 256 + threadIdx.x;       // 4,194,304
  int l = i & 1023;
  dst[i] = src[i - l + (1023 - l)];
}

__global__ void k_shuffle(const float* __restrict__ src, float* __restrict__ dst) {
  int i = blockIdx.x * 256 + threadIdx.x;       // 4,194,304
  int l = i & 1023;
  int sl = ((l & 15) << 6) | (l >> 4);
  dst[i] = src[i - l + sl];
}

// LayerNorm over C=256; fp32 in, bf16 out (feeds QKV GEMMs)
__global__ __launch_bounds__(256) void k_layernorm(const float* __restrict__ x,
                                                   const float* w, const float* b,
                                                   u16* __restrict__ o) {
  __shared__ float red[256];
  int t = threadIdx.x;
  const float* row = x + (long)blockIdx.x * 256;
  float v = row[t];
  red[t] = v; __syncthreads();
  for (int s = 128; s > 0; s >>= 1) { if (t < s) red[t] += red[t + s]; __syncthreads(); }
  float mean = red[0] * (1.f / 256.f);
  __syncthreads();
  float d = v - mean;
  red[t] = d * d; __syncthreads();
  for (int s = 128; s > 0; s >>= 1) { if (t < s) red[t] += red[t + s]; __syncthreads(); }
  float var = red[0] * (1.f / 256.f);
  o[(long)blockIdx.x * 256 + t] = f2bf(d * rsqrtf(var + 1e-5f) * w[t] + b[t]);
}

// scaled softmax over rows of 1024; fp32 scores in, bf16 probs out
__global__ __launch_bounds__(256) void k_softmax(const float* __restrict__ s,
                                                 u16* __restrict__ o) {
  __shared__ float red[256];
  const float scale = 0.08838834764831845f;     // 1/sqrt(128)
  int t = threadIdx.x;
  const float* p = s + (long)blockIdx.x * 1024;
  u16* q = o + (long)blockIdx.x * 1024;
  float v[4]; float mx = -3.0e38f;
#pragma unroll
  for (int j = 0; j < 4; ++j) { v[j] = p[t + j * 256] * scale; mx = fmaxf(mx, v[j]); }
  red[t] = mx; __syncthreads();
  for (int sw = 128; sw > 0; sw >>= 1) { if (t < sw) red[t] = fmaxf(red[t], red[t + sw]); __syncthreads(); }
  mx = red[0]; __syncthreads();
  float sum = 0.f;
#pragma unroll
  for (int j = 0; j < 4; ++j) { v[j] = __expf(v[j] - mx); sum += v[j]; }
  red[t] = sum; __syncthreads();
  for (int sw = 128; sw > 0; sw >>= 1) { if (t < sw) red[t] += red[t + sw]; __syncthreads(); }
  float inv = 1.f / red[0];
#pragma unroll
  for (int j = 0; j < 4; ++j) q[t + j * 256] = f2bf(v[j] * inv);
}

// causal depthwise conv1d (k=4) + SiLU; fp32 xm (scan) + bf16 token-major xm_t
__global__ void k_cconv(const float* __restrict__ xz, const float* __restrict__ cw,
                        const float* __restrict__ cb, float* __restrict__ out,
                        u16* __restrict__ xmt) {
  int i = blockIdx.x * 256 + threadIdx.x;       // 2,097,152 = B*512*1024
  int b = i >> 19, ch = (i >> 10) & 511, l = i & 1023;
  const float* row = xz + ((long)b << 20) + ((long)ch << 10);
  float acc = cb[ch];
#pragma unroll
  for (int j = 0; j < 4; ++j) {
    int t = l - 3 + j;
    if (t >= 0) acc += cw[ch * 4 + j] * row[t];
  }
  float s = acc / (1.f + __expf(-acc));
  out[i] = s;
  xmt[((long)b << 19) + ((long)l << 9) + ch] = f2bf(s);
}

// selective scan: 16 lanes per (b, ch) over DS=16, __shfl_xor reduction
__global__ __launch_bounds__(256) void k_scan(
    const float* __restrict__ u, const float* __restrict__ dt,
    const float* __restrict__ xdbl, const float* __restrict__ xz,
    const float* __restrict__ Alog, const float* __restrict__ Dv,
    float* __restrict__ y) {
  int tid = blockIdx.x * 256 + threadIdx.x;     // 32768
  int n  = tid & 15;
  int ch = (tid >> 4) & 511;
  int b  = tid >> 13;
  const long rowoff = ((long)b * 512 + ch) * 1024;
  const float* urow = u + rowoff;
  const float* drow = dt + rowoff;
  const float* zrow = xz + ((long)b << 20) + ((long)(512 + ch) << 10);
  const float* xd   = xdbl + (long)b * 49152;
  const float A  = -__expf(Alog[ch * 16 + n]);
  const float Dc = Dv[ch];
  float s = 0.f;
  for (int t = 0; t < 1024; ++t) {
    float del = drow[t];
    float uu  = urow[t];
    const float* bc = xd + t * 48;
    s = __expf(del * A) * s + (del * uu) * bc[16 + n];
    float yv = s * bc[32 + n];
    yv += __shfl_xor(yv, 1, 32);
    yv += __shfl_xor(yv, 2, 32);
    yv += __shfl_xor(yv, 4, 32);
    yv += __shfl_xor(yv, 8, 32);
    if (n == 0) {
      float zz = zrow[t];
      y[rowoff + t] = (yv + Dc * uu) * (zz / (1.f + __expf(-zz)));
    }
  }
}

// comb_t[b, l, ch] (bf16) = y1 + reverse_l(y2) + unshuffle(y3)
__global__ void k_combine(const float* __restrict__ y1, const float* __restrict__ y2,
                          const float* __restrict__ y3, u16* __restrict__ combt) {
  int i = blockIdx.x * 256 + threadIdx.x;       // 2,097,152 over (b,l,ch)
  int b = i >> 19, l = (i >> 9) & 1023, ch = i & 511;
  long rowoff = ((long)b * 512 + ch) * 1024;
  int l3 = ((l & 63) << 4) | (l >> 6);
  combt[i] = f2bf(y1[rowoff + l] + y2[rowoff + 1023 - l] + y3[rowoff + l3]);
}

// depthwise 3x3 (pad 1) + bias + residual; token-major fp32 xf in, NCHW out
__global__ void k_dwres(const float* __restrict__ xf, const float* __restrict__ dww,
                        const float* __restrict__ dwb, const float* __restrict__ x,
                        float* __restrict__ out) {
  int i = blockIdx.x * 256 + threadIdx.x;       // 1,048,576 (b,c,h,w)
  int b = i >> 18, c = (i >> 10) & 255, pix = i & 1023;
  int h = pix >> 5, w = pix & 31;
  float acc = dwb[c];
#pragma unroll
  for (int t = 0; t < 9; ++t) {
    int hh = h + t / 3 - 1, ww = w + t % 3 - 1;
    if ((unsigned)hh < 32u && (unsigned)ww < 32u)
      acc += dww[c * 9 + t] * xf[(((long)b << 10) + hh * 32 + ww) * 256 + c];
  }
  out[i] = acc + x[i];
}

// ---------------------------------------------------------------------------
// Host-side helpers
// ---------------------------------------------------------------------------
enum GVar { GV_BB_F32 = 0, GV_BB_BF16 = 1, GV_NGUARD = 2, GV_DT = 3 };

static void gemm(hipStream_t st, int var, const u16* A, const void* B,
                 const float* bias, void* C, int M, int N, int K,
                 long sam, long sbn, long scm, long scn,
                 int batch, int zdiv, long bA1, long bA2, long bB1, long bB2,
                 long bC1, long bC2, long bBi1, long bBi2,
                 int biasMode, int act) {
  GemmP p{A, B, bias, C, M, N, K, sam, sbn, scm, scn,
          zdiv, bA1, bA2, bB1, bB2, bC1, bC2, bBi1, bBi2, biasMode, act};
  dim3 g((N + 63) / 64, (M + 63) / 64, batch), blk(32, 8, 1);
  switch (var) {
    case GV_BB_F32:  gemm_wmma<true,  false, false, false><<<g, blk, 0, st>>>(p); break;
    case GV_BB_BF16: gemm_wmma<true,  true,  false, false><<<g, blk, 0, st>>>(p); break;
    case GV_NGUARD:  gemm_wmma<true,  false, true,  false><<<g, blk, 0, st>>>(p); break;
    default:         gemm_wmma<false, false, false, true ><<<g, blk, 0, st>>>(p); break;
  }
}

static void cvt(hipStream_t st, const float* s, u16* d, int n) {
  k_f2bfv<<<(n + 255) / 256, 256, 0, st>>>(s, d, n);
}

static void mamba_inner(hipStream_t st, const float* xzp,
                        const float* cwp, const float* cbp, const u16* xpwb,
                        const u16* dtwb, const float* dtbp,
                        const float* Alogp, const float* Dp,
                        float* xm_conv, u16* xm_tb, float* xdbl, float* dtbuf,
                        float* y) {
  k_cconv<<<8192, 256, 0, st>>>(xzp, cwp, cbp, xm_conv, xm_tb);
  // x_dbl[b,l,e] = sum_d xm_t[b,l,d]*xpw[e,d]      M=1024 N=48 K=512 -> f32
  gemm(st, GV_NGUARD, xm_tb, xpwb, nullptr, xdbl, 1024, 48, 512,
       512, 512, 48, 1,
       4, 1, 524288, 0, 0, 0, 49152, 0, 0, 0, 0, 0);
  // dt[b,d,l] = softplus(dtw @ x_dblᵀ + dtb)       M=512 N=1024 K=16 -> f32
  gemm(st, GV_DT, dtwb, xdbl, dtbp, dtbuf, 512, 1024, 16,
       16, 48, 1024, 1,
       4, 1, 0, 0, 49152, 0, 524288, 0, 0, 0, 2, 2);
  k_scan<<<128, 256, 0, st>>>(xm_conv, dtbuf, xdbl, xzp, Alogp, Dp, y);
}

extern "C" void kernel_launch(void* const* d_in, const int* in_sizes, int n_in,
                              void* d_out, int out_size, void* d_ws, size_t ws_size,
                              hipStream_t stream) {
  (void)in_sizes; (void)n_in; (void)out_size; (void)ws_size;
  const float* x       = (const float*)d_in[0];
  const float* proj1_w = (const float*)d_in[1];
  const float* proj1_b = (const float*)d_in[2];
  const float* proj2_w = (const float*)d_in[3];
  const float* proj2_b = (const float*)d_in[4];
  const float* norm_w  = (const float*)d_in[5];
  const float* norm_b  = (const float*)d_in[6];
  const float* q_w = (const float*)d_in[7];  const float* q_b = (const float*)d_in[8];
  const float* k_w = (const float*)d_in[9];  const float* k_b = (const float*)d_in[10];
  const float* v_w = (const float*)d_in[11]; const float* v_b = (const float*)d_in[12];
  const float* o_w = (const float*)d_in[13]; const float* o_b = (const float*)d_in[14];
  const float* fc1_w = (const float*)d_in[15]; const float* fc1_b = (const float*)d_in[16];
  const float* dw_w = (const float*)d_in[17]; const float* dw_b = (const float*)d_in[18];
  const float* in_w = (const float*)d_in[19];
  const float* cw  = (const float*)d_in[20]; const float* cb  = (const float*)d_in[21];
  const float* cbw = (const float*)d_in[22]; const float* cbb = (const float*)d_in[23];
  const float* csw = (const float*)d_in[24]; const float* csb = (const float*)d_in[25];
  const float* xpw  = (const float*)d_in[26];
  const float* xpbw = (const float*)d_in[27];
  const float* xpsw = (const float*)d_in[28];
  const float* dtw  = (const float*)d_in[29]; const float* dtb  = (const float*)d_in[30];
  const float* dtbw = (const float*)d_in[31]; const float* dtbb = (const float*)d_in[32];
  const float* dtsw = (const float*)d_in[33]; const float* dtsb = (const float*)d_in[34];
  const float* Alog  = (const float*)d_in[35];
  const float* Ablog = (const float*)d_in[36];
  const float* Aslog = (const float*)d_in[37];
  const float* Dv  = (const float*)d_in[38];
  const float* Dbv = (const float*)d_in[39];
  const float* Dsv = (const float*)d_in[40];
  const float* outw = (const float*)d_in[41];

  char* ws = (char*)d_ws;
  const size_t MB = 1u << 20;
  auto F  = [&](size_t mb) { return (float*)(ws + mb * MB); };
  auto Hw = [&](size_t mb) { return (u16*)(ws + mb * MB); };
  // workspace layout with reuse (peak ~113 MB)
  u16*   x_tokb = Hw(0);    u16* outmb = Hw(0);     // bf16, 2MB
  u16*   xcb    = Hw(2);    u16* xm2b  = Hw(2);     // bf16, 2MB
  float* xc     = F(4);                             // f32, 4MB (dead after LN)
  u16*   xm3b   = Hw(4);                            // reuses xc region
  u16*   Qb  = Hw(8);   u16* Kb = Hw(10);           // bf16, 2MB each
  u16*   Vtb = Hw(12);  u16* attb = Hw(14);
  float* atto = F(16);                              // f32, 4MB
  float* scores = F(20);                            // f32, 32MB
  float* xzb = F(20);   float* xzv = F(36);         // reuse scores region
  u16*   probsb = Hw(52);                           // bf16, 16MB
  u16*   hsb    = Hw(52);                           // after attention
  u16*   xm_tb  = Hw(54);                           // bf16, 4MB
  float* dtbuf  = F(58);                            // f32, 8MB
  float* xm_conv = F(68);                           // f32, 8MB
  float* xdbl    = F(76);                           // f32, 1MB
  float* y1 = F(77); float* y2 = F(85); float* y3 = F(93);
  u16*   combt = Hw(101);                           // bf16, 4MB
  float* xf    = F(105);                            // f32, 4MB
  u16*   wrepb = Hw(109);                           // bf16, 1.125MB
  u16* wb = Hw(111);                                // converted weights pool
  u16 *qwb = wb, *kwb = wb + 65536, *vwb = wb + 131072, *owb = wb + 196608;
  u16 *p2wb = wb + 262144, *fc1wb = wb + 327680;
  u16 *inwb = wb + 393216;                          // 262144
  u16 *xpwb = wb + 655360, *xpbwb = wb + 679936, *xpswb = wb + 704512;
  u16 *dtwb = wb + 729088, *dtbwb = wb + 737280, *dtswb = wb + 745472;
  u16 *outwb = wb + 753664;                         // 131072

  // 0) one-time weight conversions (bf16)
  k_wrepack<<<2304, 256, 0, stream>>>(proj1_w, wrepb);
  cvt(stream, q_w, qwb, 65536);   cvt(stream, k_w, kwb, 65536);
  cvt(stream, v_w, vwb, 65536);   cvt(stream, o_w, owb, 65536);
  cvt(stream, proj2_w, p2wb, 65536); cvt(stream, fc1_w, fc1wb, 65536);
  cvt(stream, in_w, inwb, 262144);
  cvt(stream, xpw, xpwb, 24576); cvt(stream, xpbw, xpbwb, 24576);
  cvt(stream, xpsw, xpswb, 24576);
  cvt(stream, dtw, dtwb, 8192); cvt(stream, dtbw, dtbwb, 8192);
  cvt(stream, dtsw, dtswb, 8192);
  cvt(stream, outw, outwb, 131072);

  // 1) scrambling permute (bf16 token-major) + conv1 -> f32 xc
  k_permute_in<<<4096, 256, 0, stream>>>(x, x_tokb);
  {
    ConvP c{x_tokb, wrepb, proj1_b, xc, 0};
    conv3_wmma<false><<<dim3(4, 64, 1), dim3(32, 8, 1), 0, stream>>>(c);
  }
  // 2) LayerNorm -> bf16 xcb
  k_layernorm<<<4096, 256, 0, stream>>>(xc, norm_w, norm_b, xcb);
  // 3) Q, K (bf16 out, token-major); V stored transposed per head (b,h,d,kx)
  gemm(stream, GV_BB_BF16, xcb, qwb, q_b, Qb, 4096, 256, 256, 256, 256, 256, 1,
       1, 1, 0, 0, 0, 0, 0, 0, 0, 0, 1, 0);
  gemm(stream, GV_BB_BF16, xcb, kwb, k_b, Kb, 4096, 256, 256, 256, 256, 256, 1,
       1, 1, 0, 0, 0, 0, 0, 0, 0, 0, 1, 0);
  gemm(stream, GV_BB_BF16, xcb, vwb, v_b, Vtb, 1024, 128, 256, 256, 256, 1, 1024,
       8, 2, 262144, 0, 0, 32768, 262144, 131072, 0, 128, 1, 0);
  // 4) scores = Q Kᵀ (8 head-GEMMs) -> f32
  gemm(stream, GV_BB_F32, Qb, Kb, nullptr, scores, 1024, 1024, 128,
       256, 256, 1024, 1,
       8, 2, 262144, 128, 262144, 128, 2097152, 1048576, 0, 0, 0, 0);
  // 5) softmax -> bf16 probs
  k_softmax<<<8192, 256, 0, stream>>>(scores, probsb);
  // 6) att = P Vᵀᵀ -> bf16 token-major
  gemm(stream, GV_BB_BF16, probsb, Vtb, nullptr, attb, 1024, 128, 1024,
       1024, 1024, 256, 1,
       8, 2, 1048576, 0, 262144, 131072, 262144, 128, 0, 0, 0, 0);
  // 7) output projection -> f32 atto
  gemm(stream, GV_BB_F32, attb, owb, o_b, atto, 4096, 256, 256, 256, 256, 256, 1,
       1, 1, 0, 0, 0, 0, 0, 0, 0, 0, 1, 0);
  // 8) scramble -> bf16 hs (b,l,d)
  k_scramble<<<4096, 256, 0, stream>>>(atto, hsb);
  // 9) xz[b,e,l] = in_w @ hsᵀ -> f32
  gemm(stream, GV_BB_F32, inwb, hsb, nullptr, xzb, 1024, 1024, 256,
       256, 256, 1024, 1,
       4, 1, 0, 0, 262144, 0, 1048576, 0, 0, 0, 0, 0);
  // 10) three mamba passes
  mamba_inner(stream, xzb, cw, cb, xpwb, dtwb, dtb, Alog, Dv,
              xm_conv, xm_tb, xdbl, dtbuf, y1);
  k_reverse<<<16384, 256, 0, stream>>>(xzb, xzv);
  mamba_inner(stream, xzv, cbw, cbb, xpbwb, dtbwb, dtbb, Ablog, Dbv,
              xm_conv, xm_tb, xdbl, dtbuf, y2);
  k_shuffle<<<16384, 256, 0, stream>>>(xzb, xzv);
  mamba_inner(stream, xzv, csw, csb, xpswb, dtswb, dtsb, Aslog, Dsv,
              xm_conv, xm_tb, xdbl, dtbuf, y3);
  // 11) combine -> bf16 token-major
  k_combine<<<8192, 256, 0, stream>>>(y1, y2, y3, combt);
  // 12) out_mamba = comb_t @ outwᵀ -> bf16 token-major
  gemm(stream, GV_BB_BF16, combt, outwb, nullptr, outmb, 1024, 256, 512,
       512, 512, 256, 1,
       4, 1, 524288, 0, 0, 0, 262144, 0, 0, 0, 0, 0);
  // 13) relu(conv3x3 proj1) -> bf16
  {
    ConvP c{outmb, wrepb, proj1_b, xm2b, 1};
    conv3_wmma<true><<<dim3(4, 64, 1), dim3(32, 8, 1), 0, stream>>>(c);
  }
  // 14) relu(1x1 conv proj2) -> bf16
  gemm(stream, GV_BB_BF16, xm2b, p2wb, proj2_b, xm3b, 4096, 256, 256,
       256, 256, 256, 1, 1, 1, 0, 0, 0, 0, 0, 0, 0, 0, 1, 1);
  // 15) fc1 -> f32 xf
  gemm(stream, GV_BB_F32, xm3b, fc1wb, fc1_b, xf, 4096, 256, 256,
       256, 256, 256, 1, 1, 1, 0, 0, 0, 0, 0, 0, 0, 0, 1, 0);
  // 16) depthwise 3x3 + bias + residual -> d_out
  k_dwres<<<4096, 256, 0, stream>>>(xf, dw_w, dw_b, x, (float*)d_out);
}
